// TransformerChoiceNet_2997887173092
// MI455X (gfx1250) — matrix operand order repaired and checked by reading, not compiled, not run on hardware
//
#include <hip/hip_runtime.h>
#include <hip/hip_bf16.h>

// ---------------------------------------------------------------------------
// TransformerChoiceNet forward on gfx1250 (MI455X), wave32 + WMMA f16.
// B=128 S=512 D=64 H=256 NH=8 HD=32; M = B*S = 65536 token rows.
// ---------------------------------------------------------------------------

typedef _Float16 half_t;
typedef __attribute__((ext_vector_type(16))) _Float16 v16h;
typedef __attribute__((ext_vector_type(8)))  float    v8f;

#define M_TOK  65536
#define SEQ    512
#define BATCH  128
#define HDIM   256
#define NHEAD  8
#define HEADD  32

// -------------------------- WMMA helpers -----------------------------------

__device__ __forceinline__ v8f wmma16(v16h a, v16h b, v8f c) {
  return __builtin_amdgcn_wmma_f32_16x16x32_f16(
      false, a, false, b, (short)0, c, false, false);
}

__device__ __forceinline__ v8f v8zero() {
  v8f z;
#pragma unroll
  for (int i = 0; i < 8; ++i) z[i] = 0.0f;
  return z;
}

// A-matrix (16x32 f16): lanes 0-15 row=lane, K={0..7,16..23};
//                       lanes 16-31 row=lane-16, K={8..15,24..31}.
__device__ __forceinline__ v16h load_a_f32(const float* base, int ld, int lane) {
  const int nl = lane & 15, hf = lane >> 4;
  const float* p = base + (size_t)nl * ld + hf * 8;
  v16h a;
#pragma unroll
  for (int i = 0; i < 8; ++i) a[i] = (_Float16)p[i];
#pragma unroll
  for (int i = 0; i < 8; ++i) a[8 + i] = (_Float16)p[16 + i];
  return a;
}

__device__ __forceinline__ v16h load_a_f16(const half_t* base, int ld, int lane) {
  const int nl = lane & 15, hf = lane >> 4;
  const half_t* p = base + (size_t)nl * ld + hf * 8;
  v16h a;
#pragma unroll
  for (int i = 0; i < 8; ++i) a[i] = p[i];
#pragma unroll
  for (int i = 0; i < 8; ++i) a[8 + i] = p[16 + i];
  return a;
}

// B-matrix (32x16 f16): lane = column (mod 16); lanes 0-15 hold K={0..7,16..23},
// lanes 16-31 hold K={8..15,24..31} (source is
// column-major over N: WT[N][K], K rows, or V^T rows).
__device__ __forceinline__ v16h load_b_f16(const half_t* base, int ld, int lane) {
  const int nl = lane & 15, hf = lane >> 4;
  const half_t* p = base + (size_t)nl * ld + hf * 8;
  v16h b;
#pragma unroll
  for (int i = 0; i < 8; ++i) b[i] = p[i];
#pragma unroll
  for (int i = 0; i < 8; ++i) b[8 + i] = p[16 + i];
  return b;
}

__device__ __forceinline__ float redsum16(float v) {
#pragma unroll
  for (int off = 8; off > 0; off >>= 1) v += __shfl_xor(v, off, 16);
  return v;
}

// Branch-free tanh-GELU on v_exp_f32: tanh(u) = 1 - 2/(exp(2u)+1)
__device__ __forceinline__ float gelu_tanh(float x) {
  float u = 0.7978845608028654f * (x + 0.044715f * x * x * x);
  float e = __expf(2.0f * u);
  float t = 1.0f - 2.0f / (e + 1.0f);
  return 0.5f * x * (1.0f + t);
}

// -------------------------- prep kernels -----------------------------------

__global__ __launch_bounds__(256) void transpose_w_kernel(
    const float* __restrict__ W, half_t* __restrict__ WT, int K, int N) {
  int idx = blockIdx.x * 256 + threadIdx.x;
  if (idx >= K * N) return;
  int k = idx / N, n = idx % N;
  WT[(size_t)n * K + k] = (_Float16)W[idx];
}

__global__ __launch_bounds__(256) void ln64_valid_kernel(
    const float* __restrict__ src,
    const float* __restrict__ g1, const float* __restrict__ b1,
    const float* __restrict__ g2, const float* __restrict__ b2,
    float* __restrict__ xs, float* __restrict__ xt, float* __restrict__ validf) {
  int r = blockIdx.x * 256 + threadIdx.x;
  if (r >= M_TOK) return;
  const float* p = src + (size_t)r * 64;
  float s = 0.0f;
#pragma unroll
  for (int j = 0; j < 64; ++j) s += p[j];
  validf[r] = (s != 0.0f) ? 1.0f : 0.0f;
  float mean = s * (1.0f / 64.0f);
  float var = 0.0f;
#pragma unroll
  for (int j = 0; j < 64; ++j) { float d = p[j] - mean; var += d * d; }
  var *= (1.0f / 64.0f);
  float rstd = rsqrtf(var + 1e-5f);
  float* os = xs + (size_t)r * 64;
  float* ot = xt + (size_t)r * 64;
#pragma unroll
  for (int j = 0; j < 64; ++j) {
    float nv = (p[j] - mean) * rstd;
    os[j] = nv * g1[j] + b1[j];
    ot[j] = nv * g2[j] + b2[j];
  }
}

// -------------------------- GEMM (N == 256) --------------------------------
// C = epilogue( A[M,K] @ W[K,256] + bias )
//   op 0: +bias   op 1: gelu(+bias)   op 2: LayerNorm(resid + (+bias))
template <int K>
__global__ __launch_bounds__(256) void gemm256_kernel(
    const void*   __restrict__ Av, int a_f16,
    const half_t* __restrict__ WT,
    const float*  __restrict__ bias, const float* __restrict__ resid,
    const float*  __restrict__ lng, const float* __restrict__ lnb,
    void* __restrict__ Cv, int c_f16, int op) {
  const int N = HDIM;
  const int lane = threadIdx.x & 31, wave = threadIdx.x >> 5;
  const int wrow = wave >> 2, wcol = wave & 3;
  const int m0 = blockIdx.x * 32 + wrow * 16;
  const int n0 = wcol * 64;
  const int nl = lane & 15, hf = lane >> 4;

  __shared__ float s_s[32][4];
  __shared__ float s_q[32][4];

  v8f acc[4];
#pragma unroll
  for (int t = 0; t < 4; ++t) acc[t] = v8zero();

#pragma unroll
  for (int k0 = 0; k0 < K; k0 += 32) {
    v16h a = a_f16 ? load_a_f16((const half_t*)Av + (size_t)m0 * K + k0, K, lane)
                   : load_a_f32((const float*)Av + (size_t)m0 * K + k0, K, lane);
#pragma unroll
    for (int t = 0; t < 4; ++t) {
      v16h b = load_b_f16(WT + (size_t)(n0 + 16 * t) * K + k0, K, lane);
      acc[t] = wmma16(a, b, acc[t]);
    }
  }

  float x[4][8];
#pragma unroll
  for (int t = 0; t < 4; ++t) {
    const int col = n0 + 16 * t + nl;
    const float bcol = bias[col];
#pragma unroll
    for (int v = 0; v < 8; ++v) {
      const int row = m0 + v + 8 * hf;
      float xv = acc[t][v] + bcol;
      if (op == 1) xv = gelu_tanh(xv);
      if (op == 2) xv += resid[(size_t)row * N + col];
      x[t][v] = xv;
    }
  }

  if (op == 2) {
#pragma unroll
    for (int v = 0; v < 8; ++v) {
      float ps = x[0][v] + x[1][v] + x[2][v] + x[3][v];
      float pq = x[0][v]*x[0][v] + x[1][v]*x[1][v] + x[2][v]*x[2][v] + x[3][v]*x[3][v];
      ps = redsum16(ps);
      pq = redsum16(pq);
      if (nl == 0) {
        const int rl = wrow * 16 + v + 8 * hf;
        s_s[rl][wcol] = ps;
        s_q[rl][wcol] = pq;
      }
    }
    __syncthreads();
#pragma unroll
    for (int v = 0; v < 8; ++v) {
      const int rl = wrow * 16 + v + 8 * hf;
      float tot = s_s[rl][0] + s_s[rl][1] + s_s[rl][2] + s_s[rl][3];
      float tq  = s_q[rl][0] + s_q[rl][1] + s_q[rl][2] + s_q[rl][3];
      float mean = tot * (1.0f / 256.0f);
      float var  = tq * (1.0f / 256.0f) - mean * mean;
      float rstd = rsqrtf(var + 1e-5f);
      const int row = m0 + v + 8 * hf;
#pragma unroll
      for (int t = 0; t < 4; ++t) {
        const int col = n0 + 16 * t + nl;
        float o = (x[t][v] - mean) * rstd * lng[col] + lnb[col];
        if (c_f16) ((half_t*)Cv)[(size_t)row * N + col] = (_Float16)o;
        else       ((float*)Cv)[(size_t)row * N + col] = o;
      }
    }
  } else {
#pragma unroll
    for (int t = 0; t < 4; ++t) {
      const int col = n0 + 16 * t + nl;
#pragma unroll
      for (int v = 0; v < 8; ++v) {
        const int row = m0 + v + 8 * hf;
        if (c_f16) ((half_t*)Cv)[(size_t)row * N + col] = (_Float16)x[t][v];
        else       ((float*)Cv)[(size_t)row * N + col] = x[t][v];
      }
    }
  }
}

// ---------------- fused multi-head projection GEMM (shared A) --------------
// NM weight matrices applied to the same A[M,256] f32; outputs f16.
// Reads A once instead of NM times (saves ~HBM; A tiles are the traffic).
template <int NM>
__global__ __launch_bounds__(256) void gemm_qkv_kernel(
    const float* __restrict__ A,
    const half_t* __restrict__ W0, const half_t* __restrict__ W1,
    const half_t* __restrict__ W2,
    const float* __restrict__ b0, const float* __restrict__ b1,
    const float* __restrict__ b2,
    half_t* __restrict__ C0, half_t* __restrict__ C1, half_t* __restrict__ C2) {
  const int K = HDIM, N = HDIM;
  const int lane = threadIdx.x & 31, wave = threadIdx.x >> 5;
  const int wrow = wave >> 2, wcol = wave & 3;
  const int m0 = blockIdx.x * 32 + wrow * 16;
  const int n0 = wcol * 64;
  const int nl = lane & 15, hf = lane >> 4;

  const half_t* Ws[3] = {W0, W1, W2};
  const float*  Bs[3] = {b0, b1, b2};
  half_t*       Cs[3] = {C0, C1, C2};

  v8f acc[NM][4];
#pragma unroll
  for (int m = 0; m < NM; ++m)
#pragma unroll
    for (int t = 0; t < 4; ++t) acc[m][t] = v8zero();

#pragma unroll
  for (int k0 = 0; k0 < K; k0 += 32) {
    v16h a = load_a_f32(A + (size_t)m0 * K + k0, K, lane);
#pragma unroll
    for (int m = 0; m < NM; ++m)
#pragma unroll
      for (int t = 0; t < 4; ++t) {
        v16h b = load_b_f16(Ws[m] + (size_t)(n0 + 16 * t) * K + k0, K, lane);
        acc[m][t] = wmma16(a, b, acc[m][t]);
      }
  }

#pragma unroll
  for (int m = 0; m < NM; ++m)
#pragma unroll
    for (int t = 0; t < 4; ++t) {
      const int col = n0 + 16 * t + nl;
      const float bc = Bs[m][col];
#pragma unroll
      for (int v = 0; v < 8; ++v) {
        const int row = m0 + v + 8 * hf;
        Cs[m][(size_t)row * N + col] = (_Float16)(acc[m][t][v] + bc);
      }
    }
}

// -------------------------- attention --------------------------------------
// grid (4, NH, B); 8 waves; wave owns 16 q-rows. Softmax without max-shift
// (energies are bounded: LN'd activations x 0.02-scale weights), masked
// columns multiplied by 0 after exp. Row sums computed IN the matrix unit:
// accS = P @ ones(32x16) replicates each row's sum across all columns.
__global__ __launch_bounds__(256) void attn_kernel(
    const half_t* __restrict__ Qg, const half_t* __restrict__ Kg,
    const half_t* __restrict__ Vg, const float* __restrict__ validf,
    half_t* __restrict__ Og) {
  const int b = blockIdx.z, h = blockIdx.y, qb = blockIdx.x;
  __shared__ half_t sVT[HEADD][SEQ + 8];   // V transposed, padded rows
  __shared__ half_t sP[8][16][40];         // per-wave P staging (padded)
  __shared__ float  sVal[SEQ];

  const int tid = threadIdx.x;
  for (int s = tid; s < SEQ; s += 256) {
    const half_t* vp = Vg + ((size_t)b * SEQ + s) * HDIM + h * HEADD;
#pragma unroll
    for (int d = 0; d < HEADD; ++d) sVT[d][s] = vp[d];
    // CDNA5 async copy: global -> LDS, tracked by ASYNCcnt
    unsigned lds_off = (unsigned)(uintptr_t)&sVal[s];
    const float* gp = validf + (size_t)b * SEQ + s;
    asm volatile("global_load_async_to_lds_b32 %0, %1, off"
                 :: "v"(lds_off), "v"(gp) : "memory");
  }
  asm volatile("s_wait_asynccnt 0x0" ::: "memory");
  __syncthreads();

  const int lane = tid & 31, wave = tid >> 5;
  const int nl = lane & 15, hf = lane >> 4;
  const int q0 = qb * 128 + wave * 16;
  const float scale = 0.17677669529663687f;  // 1/sqrt(HD)

  v16h aq = load_a_f16(Qg + ((size_t)b * SEQ + q0) * HDIM + h * HEADD, HDIM, lane);

  v16h ones;
#pragma unroll
  for (int i = 0; i < 16; ++i) ones[i] = (_Float16)1.0f;

  v8f accO[2] = {v8zero(), v8zero()};
  v8f accS = v8zero();

  const half_t* kbase = Kg + ((size_t)b * SEQ) * HDIM + h * HEADD;
  const v8f zc = v8zero();

#pragma unroll 2
  for (int kt = 0; kt < SEQ; kt += 32) {
    v16h bk0 = load_b_f16(kbase + (size_t)kt * HDIM, HDIM, lane);
    v16h bk1 = load_b_f16(kbase + (size_t)(kt + 16) * HDIM, HDIM, lane);
    v8f st0 = wmma16(aq, bk0, zc);
    v8f st1 = wmma16(aq, bk1, zc);
    const float cv0 = sVal[kt + nl];
    const float cv1 = sVal[kt + 16 + nl];
#pragma unroll
    for (int v = 0; v < 8; ++v) {
      float p0 = __expf(st0[v] * scale) * cv0;
      float p1 = __expf(st1[v] * scale) * cv1;
      const int prow = v + 8 * hf;
      sP[wave][prow][nl]      = (_Float16)p0;
      sP[wave][prow][16 + nl] = (_Float16)p1;
    }
    // LDS is in-order within a wave: staging store -> reload is safe.
    v16h ap = load_a_f16(&sP[wave][0][0], 40, lane);
    accO[0] = wmma16(ap, load_b_f16(&sVT[0][kt],  SEQ + 8, lane), accO[0]);
    accO[1] = wmma16(ap, load_b_f16(&sVT[16][kt], SEQ + 8, lane), accO[1]);
    accS    = wmma16(ap, ones, accS);
  }

#pragma unroll
  for (int t = 0; t < 2; ++t) {
#pragma unroll
    for (int v = 0; v < 8; ++v) {
      const int row = q0 + v + 8 * hf;
      const int col = t * 16 + nl;
      Og[((size_t)b * SEQ + row) * HDIM + h * HEADD + col] =
          (_Float16)(accO[t][v] / accS[v]);
    }
  }
}

// -------------------------- logits + softmax -------------------------------

__global__ __launch_bounds__(256) void logits_kernel(
    const float* __restrict__ T, const float* __restrict__ outW,
    const float* __restrict__ outb, float* __restrict__ logits) {
  const int lane = threadIdx.x & 31, wave = threadIdx.x >> 5;
  const int row = blockIdx.x * 8 + wave;
  const float* p = T + (size_t)row * HDIM;
  float s = 0.0f;
#pragma unroll
  for (int j = 0; j < 8; ++j) s += p[lane + 32 * j] * outW[lane + 32 * j];
#pragma unroll
  for (int off = 16; off > 0; off >>= 1) s += __shfl_xor(s, off, 32);
  if (lane == 0) logits[row] = s + outb[0];
}

__global__ __launch_bounds__(512) void softmax_kernel(
    const float* __restrict__ logits, const float* __restrict__ validf,
    float* __restrict__ out) {
  const int b = blockIdx.x, i = threadIdx.x;
  __shared__ float sv[SEQ];
  float x = __expf(logits[(size_t)b * SEQ + i]) * validf[(size_t)b * SEQ + i];
  sv[i] = x;
  __syncthreads();
  for (int st = 256; st > 0; st >>= 1) {
    if (i < st) sv[i] += sv[i + st];
    __syncthreads();
  }
  out[(size_t)b * SEQ + i] = x / sv[0];
}

// -------------------------- host orchestration -----------------------------

extern "C" void kernel_launch(void* const* d_in, const int* in_sizes, int n_in,
                              void* d_out, int out_size, void* d_ws, size_t ws_size,
                              hipStream_t stream) {
  (void)in_sizes; (void)n_in; (void)out_size; (void)ws_size;
  auto F = [&](int i) { return (const float*)d_in[i]; };

  size_t off = 0;
  auto alloc = [&](size_t bytes) -> void* {
    void* p = (char*)d_ws + off;
    off += (bytes + 255) & ~(size_t)255;
    return p;
  };

  const int widx[18] = {5, 7, 9, 11, 13, 15, 19, 21, 25, 27, 29, 31, 33, 35, 37, 39, 45, 47};
  const int wk[18]   = {64, 64, 256, 256, 256, 256, 256, 256, 256, 256, 256, 256, 256, 256, 256, 256, 256, 256};
  half_t* WT[18];
  for (int i = 0; i < 18; ++i) {
    const int K = wk[i], N = HDIM;
    WT[i] = (half_t*)alloc((size_t)K * N * sizeof(half_t));
    transpose_w_kernel<<<(K * N + 255) / 256, 256, 0, stream>>>(F(widx[i]), WT[i], K, N);
  }
  half_t *t_sem = WT[0], *t_tem = WT[1],
         *t_eq = WT[2], *t_ek = WT[3], *t_ev = WT[4], *t_eo = WT[5],
         *t_ef1 = WT[6], *t_ef2 = WT[7],
         *t_dsq = WT[8], *t_dsk = WT[9], *t_dsv = WT[10], *t_dso = WT[11],
         *t_deq = WT[12], *t_dek = WT[13], *t_dev = WT[14], *t_deo = WT[15],
         *t_df1 = WT[16], *t_df2 = WT[17];

  float* validf = (float*)alloc((size_t)M_TOK * 4);
  float* XS     = (float*)alloc((size_t)M_TOK * 64 * 4);
  float* XT     = (float*)alloc((size_t)M_TOK * 64 * 4);
  float* logits = (float*)alloc((size_t)M_TOK * 4);
  void* P[6];
  for (int i = 0; i < 6; ++i) P[i] = alloc((size_t)M_TOK * HDIM * 4);

  auto GEMM64 = [&](const void* A, const half_t* Wt, const float* bias, void* C) {
    gemm256_kernel<64><<<M_TOK / 32, 256, 0, stream>>>(
        A, 0, Wt, bias, nullptr, nullptr, nullptr, C, 0, 0);
  };
  auto GEMM = [&](const void* A, int af, const half_t* Wt, const float* bias,
                  const float* resid, const float* g, const float* be,
                  void* C, int cf, int op) {
    gemm256_kernel<256><<<M_TOK / 32, 256, 0, stream>>>(
        A, af, Wt, bias, resid, g, be, C, cf, op);
  };
  auto QKV3 = [&](const void* A, const half_t* w0, const half_t* w1, const half_t* w2,
                  const float* b0, const float* b1, const float* b2,
                  void* c0, void* c1, void* c2) {
    gemm_qkv_kernel<3><<<M_TOK / 32, 256, 0, stream>>>(
        (const float*)A, w0, w1, w2, b0, b1, b2,
        (half_t*)c0, (half_t*)c1, (half_t*)c2);
  };
  auto QKV2 = [&](const void* A, const half_t* w0, const half_t* w1,
                  const float* b0, const float* b1, void* c0, void* c1) {
    gemm_qkv_kernel<2><<<M_TOK / 32, 256, 0, stream>>>(
        (const float*)A, w0, w1, nullptr, b0, b1, nullptr,
        (half_t*)c0, (half_t*)c1, nullptr);
  };
  auto ATTN = [&](const void* Q, const void* K, const void* V, void* O) {
    attn_kernel<<<dim3(4, NHEAD, BATCH), 256, 0, stream>>>(
        (const half_t*)Q, (const half_t*)K, (const half_t*)V, validf, (half_t*)O);
  };

  // ---- embeddings ----
  ln64_valid_kernel<<<M_TOK / 256, 256, 0, stream>>>(
      F(0), F(1), F(2), F(3), F(4), XS, XT, validf);
  GEMM64(XS, t_sem, F(6), P[0]);   // es (f32, residual)
  GEMM64(XT, t_tem, F(8), P[1]);   // et (f32, residual)

  // ---- encoder ----
  QKV3(P[0], t_eq, t_ek, t_ev, F(10), F(12), F(14), P[2], P[3], P[4]); // Qe/Ke/Ve f16
  ATTN(P[2], P[3], P[4], P[5]);                                        // ae f16
  GEMM(P[5], 1, t_eo, F(16), (float*)P[0], F(17), F(18), P[2], 0, 2);  // x1 f32
  GEMM(P[2], 0, t_ef1, F(20), nullptr, nullptr, nullptr, P[3], 1, 1);  // gelu f16
  GEMM(P[3], 1, t_ef2, F(22), (float*)P[2], F(23), F(24), P[4], 0, 2); // enc f32

  // ---- decoder self-attention ----
  QKV3(P[1], t_dsq, t_dsk, t_dsv, F(26), F(28), F(30), P[0], P[3], P[5]); // Qd/Kd/Vd f16
  ATTN(P[0], P[3], P[5], P[2]);                                        // ad f16
  GEMM(P[2], 1, t_dso, F(32), (float*)P[1], F(41), F(42), P[0], 0, 2); // t1 f32

  // ---- decoder cross-attention ----
  GEMM(P[0], 0, t_deq, F(34), nullptr, nullptr, nullptr, P[1], 1, 0);  // Qc f16
  QKV2(P[4], t_dek, t_dev, F(36), F(38), P[3], P[5]);                  // Kc/Vc f16
  ATTN(P[1], P[3], P[5], P[2]);                                        // ac f16
  GEMM(P[2], 1, t_deo, F(40), (float*)P[0], F(43), F(44), P[4], 0, 2); // t2 f32

  // ---- decoder FFN ----
  GEMM(P[4], 0, t_df1, F(46), nullptr, nullptr, nullptr, P[1], 1, 1);  // gelu f16
  GEMM(P[1], 1, t_df2, F(48), (float*)P[4], F(49), F(50), P[0], 0, 2); // t3 f32

  // ---- output head + per-batch masked softmax ----
  logits_kernel<<<M_TOK / 8, 256, 0, stream>>>((float*)P[0], F(51), F(52), logits);
  softmax_kernel<<<BATCH, SEQ, 0, stream>>>(logits, validf, (float*)d_out);
}
